// PLDA_15393162789004
// MI455X (gfx1250) — matrix operand order repaired and checked by reading, not compiled, not run on hardware
//
#include <hip/hip_runtime.h>
#include <hip/hip_bf16.h>
#include <math.h>

typedef __bf16 bf16_t;
typedef __attribute__((ext_vector_type(16))) __bf16 v16bf;
typedef __attribute__((ext_vector_type(8)))  float  v8f;
typedef __attribute__((ext_vector_type(2)))  float  v2f;

union FragBF16 { v16bf v; uint4 u[2]; };

#define DD 128          // feature dim
#define KC 256          // classes
#define NC 64           // per-class
#define MM 8192         // test rows
#define CNT 16384       // KC*NC

// ---------------- Stage 1: means ----------------
__global__ __launch_bounds__(128) void plda_class_mean(const float* __restrict__ src,
                                                       float* __restrict__ mpc) {
    const int k = blockIdx.x, d = threadIdx.x;
    float s = 0.0f;
    const float* p = src + ((size_t)k * NC) * DD + d;
    for (int n = 0; n < NC; ++n) s += p[(size_t)n * DD];
    mpc[k * DD + d] = s * (1.0f / NC);
}

__global__ __launch_bounds__(128) void plda_global_mean(const float* __restrict__ mpc,
                                                        float* __restrict__ mean) {
    const int d = threadIdx.x;
    float s = 0.0f;
    for (int k = 0; k < KC; ++k) s += mpc[k * DD + d];
    mean[d] = s * (1.0f / KC);
}

// ---------------- Stage 2a: S_w via fp32 WMMA (16x16x4) ----------------
// S_w[de] tile per 1-wave block: C = A*B, A[m][k]=diff_w[k][dm+m], B[k][n]=diff_w[k][de+n]
__global__ __launch_bounds__(32) void plda_scatter_w(const float* __restrict__ src,
                                                     const float* __restrict__ mpc,
                                                     float* __restrict__ Sw) {
    const int lane = threadIdx.x & 31;
    const int half = lane >> 4, l15 = lane & 15;
    const int dm = blockIdx.y * 16, de = blockIdx.x * 16;
    v8f acc = {0,0,0,0,0,0,0,0};
    for (int k0 = 0; k0 < CNT; k0 += 4) {
        const int ka = k0 + half * 2;        // A/B f32 frag: v0 -> K=half*2, v1 -> K=half*2+1
        const int kb = ka + 1;
        const int c0 = ka >> 6, c1 = kb >> 6;
        v2f a, b;
        a.x = src[(size_t)ka * DD + dm + l15] - mpc[c0 * DD + dm + l15];
        a.y = src[(size_t)kb * DD + dm + l15] - mpc[c1 * DD + dm + l15];
        b.x = src[(size_t)ka * DD + de + l15] - mpc[c0 * DD + de + l15];
        b.y = src[(size_t)kb * DD + de + l15] - mpc[c1 * DD + de + l15];
        acc = __builtin_amdgcn_wmma_f32_16x16x4_f32(false, a, false, b,
                                                    (short)0, acc, false, false);
    }
#pragma unroll
    for (int r = 0; r < 8; ++r) {
        const int row = dm + r + half * 8;
        Sw[row * DD + de + l15] = acc[r] * (1.0f / CNT);
    }
}

// ---------------- Stage 2b: S_b (tiny, plain FMA) ----------------
__global__ __launch_bounds__(128) void plda_scatter_b(const float* __restrict__ mpc,
                                                      const float* __restrict__ mean,
                                                      float* __restrict__ Sb) {
    const int d = blockIdx.x, e = threadIdx.x;
    const float md = mean[d], me = mean[e];
    float s = 0.0f;
    for (int k = 0; k < KC; ++k)
        s += (mpc[k * DD + d] - md) * (mpc[k * DD + e] - me);
    Sb[d * DD + e] = s * ((float)NC / (float)CNT);   // == s/256
}

// ---------------- Stage 3: Cholesky + Jacobi eigh + W solve ----------------
__global__ __launch_bounds__(128) void plda_fit(float* __restrict__ Sw,   // -> L (lower) in place
                                                float* __restrict__ Sb,   // -> Msym -> Jacobi M
                                                float* __restrict__ Ub,   // tmp T, then U
                                                float* __restrict__ Wb,   // tmp Y, then W
                                                float* __restrict__ Ainv,
                                                float* __restrict__ psi,
                                                float* __restrict__ logd) {
    const int tid = threadIdx.x;
    __shared__ float cs[64], sn[64];
    __shared__ int   pr[64], qr[64];
    __shared__ float red[128];

    // Cholesky of Sw (lower), in place
    for (int k = 0; k < DD; ++k) {
        if (tid == 0) Sw[k * DD + k] = sqrtf(fmaxf(Sw[k * DD + k], 1e-30f));
        __syncthreads();
        const float lkk = Sw[k * DD + k];
        const int i = k + 1 + tid;
        if (i < DD) Sw[i * DD + k] /= lkk;
        __syncthreads();
        if (i < DD) {
            const float lik = Sw[i * DD + k];
            for (int j = k + 1; j <= i; ++j) Sw[i * DD + j] -= lik * Sw[j * DD + k];
        }
        __syncthreads();
    }

    // T = L^{-1} Sb  (column tid) -> Ub
    {
        const int c = tid;
        for (int i = 0; i < DD; ++i) {
            float s = Sb[i * DD + c];
            for (int m = 0; m < i; ++m) s -= Sw[i * DD + m] * Ub[m * DD + c];
            Ub[i * DD + c] = s / Sw[i * DD + i];
        }
    }
    __syncthreads();
    // Y = L^{-1} T^T (rhs_i = T[c][i]) -> Wb
    {
        const int c = tid;
        for (int i = 0; i < DD; ++i) {
            float s = Ub[c * DD + i];
            for (int m = 0; m < i; ++m) s -= Sw[i * DD + m] * Wb[m * DD + c];
            Wb[i * DD + c] = s / Sw[i * DD + i];
        }
    }
    __syncthreads();
    // Msym = 0.5*(Y + Y^T) -> Sb ; U = I -> Ub
    for (int j = 0; j < DD; ++j) Sb[tid * DD + j] = 0.5f * (Wb[tid * DD + j] + Wb[j * DD + tid]);
    for (int j = 0; j < DD; ++j) Ub[tid * DD + j] = (tid == j) ? 1.0f : 0.0f;
    __syncthreads();

    // Cyclic Jacobi, tournament schedule (closed form), 10 sweeps
    float* Mb = Sb;
    for (int sweep = 0; sweep < 10; ++sweep) {
        for (int step = 0; step < 127; ++step) {
            if (tid < 64) {
                const int t  = tid;
                const int i1 = (t == 0) ? 0 : 1 + (t - 1 + step) % 127;
                const int i2 = 1 + (126 - t + step) % 127;
                const int p = (i1 < i2) ? i1 : i2;
                const int q = (i1 < i2) ? i2 : i1;
                const float app = Mb[p * DD + p], aqq = Mb[q * DD + q], apq = Mb[p * DD + q];
                float c = 1.0f, s = 0.0f;
                if (fabsf(apq) > 1e-12f) {
                    const float th = (aqq - app) / (2.0f * apq);
                    float tt = 1.0f / (fabsf(th) + sqrtf(1.0f + th * th));
                    if (th < 0.0f) tt = -tt;
                    c = 1.0f / sqrtf(1.0f + tt * tt);
                    s = tt * c;
                }
                cs[t] = c; sn[t] = s; pr[t] = p; qr[t] = q;
            }
            __syncthreads();
            {   // rows p,q of M
                const int t = tid >> 1, jb = (tid & 1) * 64;
                const int p = pr[t], q = qr[t];
                const float c = cs[t], s = sn[t];
                for (int j = jb; j < jb + 64; ++j) {
                    const float mp = Mb[p * DD + j], mq = Mb[q * DD + j];
                    Mb[p * DD + j] = c * mp - s * mq;
                    Mb[q * DD + j] = s * mp + c * mq;
                }
            }
            __syncthreads();
            {   // cols p,q of M and U
                const int t = tid >> 1, ib = (tid & 1) * 64;
                const int p = pr[t], q = qr[t];
                const float c = cs[t], s = sn[t];
                for (int i = ib; i < ib + 64; ++i) {
                    const float mp = Mb[i * DD + p], mq = Mb[i * DD + q];
                    Mb[i * DD + p] = c * mp - s * mq;
                    Mb[i * DD + q] = s * mp + c * mq;
                    const float up = Ub[i * DD + p], uq = Ub[i * DD + q];
                    Ub[i * DD + p] = c * up - s * uq;
                    Ub[i * DD + q] = s * up + c * uq;
                }
            }
            __syncthreads();
        }
    }

    // W = L^{-T} U  (column tid) -> Wb
    {
        const int c = tid;
        for (int i = DD - 1; i >= 0; --i) {
            float s = Ub[i * DD + c];
            for (int m = i + 1; m < DD; ++m) s -= Sw[m * DD + i] * Wb[m * DD + c];
            Wb[i * DD + c] = s / Sw[i * DD + i];
        }
    }
    __syncthreads();

    // lambda_w == 1, lambda_b == evals  (exact identities)
    {
        const float nn = (float)NC;
        const float ev = Mb[tid * DD + tid];
        const float ps = fmaxf((nn - 1.0f) / nn * ev - 1.0f / nn, 0.0f);
        psi[tid] = ps;
        red[tid] = ps * ps;
    }
    __syncthreads();
    for (int ofs = 64; ofs > 0; ofs >>= 1) {
        if (tid < ofs) red[tid] += red[tid + ofs];
        __syncthreads();
    }
    if (tid == 0)
        logd[0] = 0.5f * ((float)DD * logf(2.0f * 3.14159265358979323846f) + 0.5f * logf(red[0]));
    const float scale = sqrtf((float)(NC - 1) / (float)NC);
    for (int c2 = 0; c2 < DD; ++c2) Ainv[tid * DD + c2] = scale * Wb[c2 * DD + tid];
}

// ---------------- Stage 4: latent_s = sqrt(psi) * A^{-1}(x-mean), q = ||.||^2 ----------------
__global__ __launch_bounds__(128) void plda_latent(const float* __restrict__ test,
                                                   const float* __restrict__ mean,
                                                   const float* __restrict__ Ainv,
                                                   const float* __restrict__ psi,
                                                   bf16_t* __restrict__ lat,
                                                   float* __restrict__ qv) {
    __shared__ float xm[DD];
    __shared__ float red[DD];
    const int row = blockIdx.x, d = threadIdx.x;
    xm[d] = test[(size_t)row * DD + d] - mean[d];
    __syncthreads();
    const float* ar = Ainv + d * DD;
    float y = 0.0f;
#pragma unroll 8
    for (int e = 0; e < DD; ++e) y += ar[e] * xm[e];
    const float ls = sqrtf(psi[d]) * y;
    const bf16_t lb = (bf16_t)ls;
    lat[(size_t)row * DD + d] = lb;
    const float lsb = (float)lb;                 // use rounded value for q too
    red[d] = lsb * lsb;
    __syncthreads();
    for (int ofs = 64; ofs > 0; ofs >>= 1) {
        if (d < ofs) red[d] += red[d + ofs];
        __syncthreads();
    }
    if (d == 0) qv[row] = red[0];
}

// ---------------- Stage 5: out[i][j] = 2*G_ij - q_i - q_j - log_div (bf16 WMMA) ----------------
// 256 threads = 8 waves; block tile 128x128; wave grid 2(M) x 4(N); wave tile 64x32.
__global__ __launch_bounds__(256) void plda_pairwise(const bf16_t* __restrict__ lat,
                                                     const float* __restrict__ qv,
                                                     const float* __restrict__ logd,
                                                     float* __restrict__ out) {
    const int wid  = threadIdx.x >> 5;
    const int lane = threadIdx.x & 31;
    const int waveN = wid & 3, waveM = wid >> 2;
    const int half = lane >> 4, l15 = lane & 15;
    const int bM = blockIdx.y * 128 + waveM * 64;
    const int bN = blockIdx.x * 128 + waveN * 32;
    const float logdiv = *logd;

    v8f acc[4][2];
#pragma unroll
    for (int mi = 0; mi < 4; ++mi)
#pragma unroll
        for (int ni = 0; ni < 2; ++ni) acc[mi][ni] = (v8f){0,0,0,0,0,0,0,0};

#pragma unroll
    for (int kt = 0; kt < 4; ++kt) {
        const int k0 = kt * 32;
        FragBF16 a[4], b[2];
#pragma unroll
        for (int mi = 0; mi < 4; ++mi) {
            // A 16x32 bf16: lanes0-15 K=[0..7],[16..23]; lanes16-31 K=[8..15],[24..31]
            const bf16_t* rp = lat + (size_t)(bM + mi * 16 + l15) * DD;
            a[mi].u[0] = *(const uint4*)(rp + k0 + half * 8);
            a[mi].u[1] = *(const uint4*)(rp + k0 + 16 + half * 8);
        }
#pragma unroll
        for (int ni = 0; ni < 2; ++ni) {
            // B 32x16 bf16: lanes0-15 K=[0..7],[16..23]; lanes16-31 K=[8..15],[24..31]
            const bf16_t* rp = lat + (size_t)(bN + ni * 16 + l15) * DD;
            b[ni].u[0] = *(const uint4*)(rp + k0 + half * 8);
            b[ni].u[1] = *(const uint4*)(rp + k0 + 16 + half * 8);
        }
#pragma unroll
        for (int mi = 0; mi < 4; ++mi)
#pragma unroll
            for (int ni = 0; ni < 2; ++ni)
                acc[mi][ni] = __builtin_amdgcn_wmma_f32_16x16x32_bf16(
                    false, a[mi].v, false, b[ni].v, (short)0, acc[mi][ni], false, false);
    }

#pragma unroll
    for (int mi = 0; mi < 4; ++mi) {
        float qi[8];
#pragma unroll
        for (int r = 0; r < 8; ++r) qi[r] = qv[bM + mi * 16 + half * 8 + r];
#pragma unroll
        for (int ni = 0; ni < 2; ++ni) {
            const int j = bN + ni * 16 + l15;
            const float qj = qv[j];
#pragma unroll
            for (int r = 0; r < 8; ++r) {
                const int i = bM + mi * 16 + half * 8 + r;
                out[(size_t)i * MM + j] = 2.0f * acc[mi][ni][r] - qi[r] - qj - logdiv;
            }
        }
    }
}

// ---------------- Host launcher ----------------
extern "C" void kernel_launch(void* const* d_in, const int* in_sizes, int n_in,
                              void* d_out, int out_size, void* d_ws, size_t ws_size,
                              hipStream_t stream) {
    const float* src  = (const float*)d_in[0];   // [256,64,128]
    const float* test = (const float*)d_in[1];   // [8192,128]
    float* out = (float*)d_out;                  // [8192,8192]
    float* ws  = (float*)d_ws;

    float* mpc  = ws;             // 32768
    float* mean = ws + 32768;     // 128
    float* Sw   = ws + 32896;     // 16384
    float* Sb   = ws + 49280;     // 16384
    float* Ub   = ws + 65664;     // 16384
    float* Wb   = ws + 82048;     // 16384
    float* Ainv = ws + 98432;     // 16384
    float* psi  = ws + 114816;    // 128
    float* logd = ws + 114944;    // 8 (pad)
    float* qv   = ws + 114952;    // 8192
    bf16_t* lat = (bf16_t*)(ws + 123144); // 8192*128 bf16 (16B-aligned)

    plda_class_mean <<<KC, DD, 0, stream>>>(src, mpc);
    plda_global_mean<<<1,  DD, 0, stream>>>(mpc, mean);
    plda_scatter_w  <<<dim3(8, 8), 32, 0, stream>>>(src, mpc, Sw);
    plda_scatter_b  <<<DD, DD, 0, stream>>>(mpc, mean, Sb);
    plda_fit        <<<1, DD, 0, stream>>>(Sw, Sb, Ub, Wb, Ainv, psi, logd);
    plda_latent     <<<MM, DD, 0, stream>>>(test, mean, Ainv, psi, lat, qv);
    plda_pairwise   <<<dim3(MM / 128, MM / 128), 256, 0, stream>>>(lat, qv, logd, out);
}
